// FSE_Module_9251359556117
// MI455X (gfx1250) — compile-verified, matrix operand order checked by reading, not run on hardware
//
#include <hip/hip_runtime.h>
#include <hip/hip_bf16.h>

// ---------------------------------------------------------------------------
// Types for CDNA5 WMMA (wave32): bf16 A/B fragments, f32 accumulator.
// ---------------------------------------------------------------------------
typedef __attribute__((ext_vector_type(16))) __bf16 bf16x16;
typedef __attribute__((ext_vector_type(8)))  __bf16 bf16x8;
typedef __attribute__((ext_vector_type(8)))  float  v8f;
typedef __attribute__((ext_vector_type(8)))  unsigned short u16x8;
typedef __attribute__((ext_vector_type(4)))  int    v4i;

// Probe for the gfx1250 async global->LDS copy builtins (toolchain-dependent).
#if defined(__has_builtin)
#  if __has_builtin(__builtin_amdgcn_global_load_async_to_lds_b128) && \
      __has_builtin(__builtin_amdgcn_s_wait_asynccnt)
#    define USE_ASYNC_LDS 1
#  endif
#endif
#ifndef USE_ASYNC_LDS
#  define USE_ASYNC_LDS 0
#endif

#if USE_ASYNC_LDS
// Builtin signature (from hipcc diagnostic): param0 = v4i in global AS,
// param1 = v4i in LDS AS, then imm offset + imm cpol.
typedef __attribute__((address_space(1))) v4i* as1_v4ip;
typedef __attribute__((address_space(3))) v4i* as3_v4ip;
#endif

__device__ __forceinline__ void copy16B_to_lds(const unsigned short* g, unsigned short* l) {
#if USE_ASYNC_LDS
    __builtin_amdgcn_global_load_async_to_lds_b128(
        (as1_v4ip)(v4i*)(unsigned short*)g,   // drop const, retype, addrspacecast
        (as3_v4ip)(v4i*)l, 0, 0);
#else
    *(u16x8*)l = *(const u16x8*)g;
#endif
}

__device__ __forceinline__ void staging_wait() {
#if USE_ASYNC_LDS
    __builtin_amdgcn_s_wait_asynccnt(0);
#endif
    __syncthreads();
}

__device__ __forceinline__ unsigned short f2bf(float f) {
    unsigned int u = __float_as_uint(f);
    u += 0x7FFFu + ((u >> 16) & 1u);   // round-to-nearest-even
    return (unsigned short)(u >> 16);
}

__device__ __forceinline__ float mishf(float x) {
    float sp = (x > 20.f) ? x : log1pf(expf(x));   // softplus, overflow-safe
    return x * tanhf(sp);
}

// ---------------------------------------------------------------------------
// Implicit-GEMM conv via V_WMMA_F32_16X16X32_BF16.
//   GEMM view: M = COUT, N = batch*IH*IW pixels, K = CIN*(3*3 or 1).
//   Workgroup = 8 waves (WM2 x WN2); wave tile = 2 M-subtiles x 4 N-subtiles
//   (32 couts x 64 px, 8 f32 accumulators). Per K-chunk (32): stage A
//   (weights, async b128 when available) + B (im2col gather) in LDS,
//   K-contiguous rows, stride 40 elems (80B) -> conflict-free ds_load_b128.
//   All per-pixel staging address math is hoisted out of the K loop
//   (NT % BN == 0 so each thread's pixel is k-invariant).
// A-frag layout (16x32 bf16): lane&15 = M row; lanes>=16 hold K=8..15 / 24..31.
// B-frag symmetric with N columns. C/D: VGPR r -> M=r (+8 for lanes>=16), N=lane&15.
// ---------------------------------------------------------------------------
template<int CIN, int COUT, int KTOT, int WM2, int WN2, bool K3, int IH, int IW>
__global__ __launch_bounds__(32 * WM2 * WN2)
void conv_wmma(const unsigned short* __restrict__ src,   // bf16 NCHW activations
               const unsigned short* __restrict__ wgt,   // bf16 [COUT][KTOT]
               const float*          __restrict__ bias,  // [COUT]
               float*                __restrict__ dst)   // f32 NCHW pre-BN output
{
    constexpr int BM  = 32 * WM2;          // couts per workgroup
    constexpr int BN  = 64 * WN2;          // pixels per workgroup
    constexpr int NT  = 32 * WM2 * WN2;    // threads (256)
    constexpr int LD  = 40;                // LDS row stride (bf16 elems), 80B
    constexpr int HWp = IH * IW;
    static_assert(NT % BN == 0, "staging geometry");
    constexpr int BSTEP = NT / BN;         // k-rows advanced per staging step
    constexpr int BITER = (BN * 32) / NT;  // staged elements per thread

    __shared__ unsigned short sA[BM * LD];
    __shared__ unsigned short sB[BN * LD];

    const int  tid  = threadIdx.x;
    const int  lane = tid & 31;
    const int  wid  = tid >> 5;
    const int  wm   = wid % WM2;
    const int  wn   = wid / WM2;
    const int  l15  = lane & 15;
    const int  kh   = (lane >> 4) * 8;     // K-half select per ISA layout
    const long pixBase = (long)blockIdx.x * BN;

    // ---- hoisted B-staging geometry (k-invariant per thread) ----
    const int  px   = tid % BN;
    const int  kkB  = tid / BN;
    const long pB   = pixBase + px;
    const int  nB   = (int)(pB / HWp);
    const int  hwB  = (int)(pB - (long)nB * HWp);
    const int  hB   = hwB / IW;
    const int  wB   = hwB - hB * IW;
    const long nbase = (long)nB * CIN * HWp;
    unsigned short* sBrow = &sB[px * LD];

    v8f acc[2][4];
#pragma unroll
    for (int m = 0; m < 2; ++m)
#pragma unroll
        for (int t = 0; t < 4; ++t) acc[m][t] = {};

    for (int kc = 0; kc < KTOT / 32; ++kc) {
        const int k0 = kc * 32;

        // ---- stage A: BM x 32 weights, 16B per lane (async when available) ----
#pragma unroll
        for (int e = tid; e < BM * 4; e += NT) {
            const int row = e >> 2;
            const int c8  = (e & 3) * 8;
            copy16B_to_lds(&wgt[row * KTOT + k0 + c8], &sA[row * LD + c8]);
        }
        // ---- stage B: im2col gather, pixel-major (coalesced global reads) ----
#pragma unroll
        for (int j = 0; j < BITER; ++j) {
            const int kk = kkB + j * BSTEP;
            const int k  = k0 + kk;
            unsigned short v;
            if (K3) {
                const int cin = k / 9;
                const int r9  = k - cin * 9;
                const int ky  = r9 / 3;
                const int kx  = r9 - ky * 3;
                const int ih  = hB + ky - 1;
                const int iw  = wB + kx - 1;
                const bool ok = ((unsigned)ih < (unsigned)IH) & ((unsigned)iw < (unsigned)IW);
                v = ok ? src[nbase + (long)cin * HWp + (long)ih * IW + iw]
                       : (unsigned short)0;
            } else {
                v = src[nbase + (long)k * HWp + hwB];
            }
            sBrow[kk] = v;
        }
        staging_wait();

        // ---- fragments + 8 WMMAs (B reused across the 2 M-subtiles) ----
        bf16x16 afrag[2];
#pragma unroll
        for (int m = 0; m < 2; ++m) {
            const unsigned short* pa = &sA[((wm * 2 + m) * 16 + l15) * LD];
            bf16x8 alo = *(const bf16x8*)(pa + kh);
            bf16x8 ahi = *(const bf16x8*)(pa + 16 + kh);
#pragma unroll
            for (int i = 0; i < 8; ++i) { afrag[m][i] = alo[i]; afrag[m][i + 8] = ahi[i]; }
        }
#pragma unroll
        for (int t = 0; t < 4; ++t) {
            const unsigned short* pb = &sB[(wn * 64 + t * 16 + l15) * LD];
            bf16x8 blo = *(const bf16x8*)(pb + kh);
            bf16x8 bhi = *(const bf16x8*)(pb + 16 + kh);
            bf16x16 bfrag;
#pragma unroll
            for (int i = 0; i < 8; ++i) { bfrag[i] = blo[i]; bfrag[i + 8] = bhi[i]; }
#pragma unroll
            for (int m = 0; m < 2; ++m) {
                acc[m][t] = __builtin_amdgcn_wmma_f32_16x16x32_bf16(
                    false, afrag[m], false, bfrag, (short)0, acc[m][t], false, false);
            }
        }
        __syncthreads();
    }

    // ---- epilogue: + conv bias, scatter f32 NCHW (coalesced along W per row) ----
#pragma unroll
    for (int t = 0; t < 4; ++t) {
        const long p  = pixBase + wn * 64 + t * 16 + l15;
        const int  n  = (int)(p / HWp);
        const int  hw = (int)(p - (long)n * HWp);
#pragma unroll
        for (int m = 0; m < 2; ++m) {
            const int coutBase = (wm * 2 + m) * 16 + ((lane >> 4) * 8);
#pragma unroll
            for (int r = 0; r < 8; ++r) {
                const int co = coutBase + r;
                dst[((long)n * COUT + co) * HWp + hw] = acc[m][t][r] + bias[co];
            }
        }
    }
}

// ---------------------------------------------------------------------------
// BatchNorm stats: per-channel sum / sumsq via block reduction + f32 atomics.
// ---------------------------------------------------------------------------
__global__ void bn_stats(const float* __restrict__ x, float* __restrict__ sum,
                         float* __restrict__ sumsq, int C, int HWt,
                         long perChan, int slabs)
{
    __shared__ float rs[256];
    __shared__ float rq[256];
    const int c    = blockIdx.x % C;
    const int slab = blockIdx.x / C;
    float s = 0.f, q = 0.f;
    const long stride = (long)slabs * 256;
    for (long i = (long)slab * 256 + threadIdx.x; i < perChan; i += stride) {
        const int n = (int)(i / HWt);
        const int r = (int)(i - (long)n * HWt);
        const float v = x[((long)n * C + c) * HWt + r];
        s += v; q += v * v;
    }
    rs[threadIdx.x] = s; rq[threadIdx.x] = q;
    __syncthreads();
    for (int o = 128; o > 0; o >>= 1) {
        if (threadIdx.x < o) {
            rs[threadIdx.x] += rs[threadIdx.x + o];
            rq[threadIdx.x] += rq[threadIdx.x + o];
        }
        __syncthreads();
    }
    if (threadIdx.x == 0) {
        atomicAdd(&sum[c], rs[0]);
        atomicAdd(&sumsq[c], rq[0]);
    }
}

__global__ void bn_finalize(const float* __restrict__ sum, const float* __restrict__ sumsq,
                            const float* __restrict__ gamma, const float* __restrict__ beta,
                            float* __restrict__ scale, float* __restrict__ shift,
                            int C, float invCnt)
{
    const int c = blockIdx.x * blockDim.x + threadIdx.x;
    if (c < C) {
        const float m = sum[c] * invCnt;
        const float v = sumsq[c] * invCnt - m * m;
        const float s = gamma[c] * rsqrtf(v + 1e-5f);
        scale[c] = s;
        shift[c] = beta[c] - m * s;
    }
}

// BN + mish, emitting bf16 (next conv's input)
__global__ void bn_mish_bf16(const float* __restrict__ x, const float* __restrict__ scale,
                             const float* __restrict__ shift, unsigned short* __restrict__ out,
                             int C, int HWt, long total)
{
    for (long i = blockIdx.x * 256L + threadIdx.x; i < total; i += gridDim.x * 256L) {
        const int c = (int)((i / HWt) % C);
        out[i] = f2bf(mishf(x[i] * scale[c] + shift[c]));
    }
}

// BN + mish, emitting f32 (final x_high)
__global__ void bn_mish_f32(const float* __restrict__ x, const float* __restrict__ scale,
                            const float* __restrict__ shift, float* __restrict__ out,
                            int C, int HWt, long total)
{
    for (long i = blockIdx.x * 256L + threadIdx.x; i < total; i += gridDim.x * 256L) {
        const int c = (int)((i / HWt) % C);
        out[i] = mishf(x[i] * scale[c] + shift[c]);
    }
}

__global__ void cvt_bf16(const float* __restrict__ in, unsigned short* __restrict__ out, long n)
{
    for (long i = blockIdx.x * 256L + threadIdx.x; i < n; i += gridDim.x * 256L)
        out[i] = f2bf(in[i]);
}

// ---------------------------------------------------------------------------
// Fused: y = mish(bn2(h2)) + x  ->  Haar DWT.
//   cA -> d_out (x_low, f32); cH/cV/cD -> bf16 (8,192,128,128) for 1x1 conv.
// ---------------------------------------------------------------------------
__global__ void fuse_residual_haar(const float* __restrict__ h2, const float* __restrict__ xres,
                                   const float* __restrict__ scale, const float* __restrict__ shift,
                                   float* __restrict__ outLow, unsigned short* __restrict__ high)
{
    const long total = 8L * 64 * 128 * 128;
    for (long idx = blockIdx.x * 256L + threadIdx.x; idx < total; idx += gridDim.x * 256L) {
        const int ow = (int)(idx & 127);
        const int oh = (int)((idx >> 7) & 127);
        const int c  = (int)((idx >> 14) & 63);
        const int n  = (int)(idx >> 20);
        const long base = (((long)n * 64 + c) * 256 + oh * 2) * 256 + ow * 2;
        const float s = scale[c], sh = shift[c];
        const float a  = mishf(h2[base]       * s + sh) + xres[base];
        const float b  = mishf(h2[base + 1]   * s + sh) + xres[base + 1];
        const float cc = mishf(h2[base + 256] * s + sh) + xres[base + 256];
        const float d  = mishf(h2[base + 257] * s + sh) + xres[base + 257];
        outLow[idx] = (a + b + cc + d) * 0.5f;
        const long hb = (((long)n * 192 + c) * 128 + oh) * 128 + ow;
        high[hb]                = f2bf((a + b - cc - d) * 0.5f);  // cH -> ch c
        high[hb + 64L * 16384]  = f2bf((a - b + cc - d) * 0.5f);  // cV -> ch c+64
        high[hb + 128L * 16384] = f2bf((a - b - cc + d) * 0.5f);  // cD -> ch c+128
    }
}

// ---------------------------------------------------------------------------
// Orchestration. Workspace layout (bytes, 256-aligned):
//   0         stats (12 x 128 f32)
//   8192      w1 bf16  [128][576]   (147456)
//   155648    w2 bf16  [64][1152]   (147456)
//   303104    wh bf16  [64][192]    (24576)
//   327680    x  bf16  (8,64,256,256)    (64 MiB)
//   67436544  h1 pre-BN f32 (8,128,256,256) (256 MiB)
//   335872000 h1 bf16       (8,128,256,256) (128 MiB)
//   470089728 h2 pre-BN f32 (8,64,256,256)  (128 MiB)
//   604307456 high bf16     (8,192,128,128) (48 MiB)
//   654639104 hh pre-BN f32 (8,64,128,128)  (32 MiB)
// ---------------------------------------------------------------------------
extern "C" void kernel_launch(void* const* d_in, const int* in_sizes, int n_in,
                              void* d_out, int out_size, void* d_ws, size_t ws_size,
                              hipStream_t stream)
{
    (void)in_sizes; (void)n_in; (void)out_size; (void)ws_size;
    const float* x   = (const float*)d_in[0];
    const float* w1  = (const float*)d_in[1];
    const float* b1  = (const float*)d_in[2];
    const float* g1  = (const float*)d_in[3];
    const float* be1 = (const float*)d_in[4];
    const float* w2  = (const float*)d_in[5];
    const float* b2  = (const float*)d_in[6];
    const float* g2  = (const float*)d_in[7];
    const float* be2 = (const float*)d_in[8];
    const float* wh  = (const float*)d_in[9];
    const float* bh  = (const float*)d_in[10];
    const float* gh  = (const float*)d_in[11];
    const float* beh = (const float*)d_in[12];

    char*  ws    = (char*)d_ws;
    float* stats = (float*)ws;
    float* sum1 = stats;          float* sq1 = stats + 128;
    float* sc1  = stats + 256;    float* sh1 = stats + 384;
    float* sum2 = stats + 512;    float* sq2 = stats + 640;
    float* sc2  = stats + 768;    float* sh2 = stats + 896;
    float* sumh = stats + 1024;   float* sqh = stats + 1152;
    float* sch  = stats + 1280;   float* shh = stats + 1408;

    unsigned short* w1b = (unsigned short*)(ws + 8192);
    unsigned short* w2b = (unsigned short*)(ws + 155648);
    unsigned short* whb = (unsigned short*)(ws + 303104);
    unsigned short* xb  = (unsigned short*)(ws + 327680);
    float*          h1p = (float*)(ws + 67436544);
    unsigned short* h1b = (unsigned short*)(ws + 335872000);
    float*          h2p = (float*)(ws + 470089728);
    unsigned short* hib = (unsigned short*)(ws + 604307456);
    float*          hhp = (float*)(ws + 654639104);

    (void)hipMemsetAsync(stats, 0, 1536 * sizeof(float), stream);

    // weight + input conversions to bf16 (OIHW flat order == GEMM [cout][k])
    cvt_bf16<<<256, 256, 0, stream>>>(w1, w1b, 128L * 64 * 9);
    cvt_bf16<<<256, 256, 0, stream>>>(w2, w2b, 64L * 128 * 9);
    cvt_bf16<<<64,  256, 0, stream>>>(wh, whb, 64L * 192);
    cvt_bf16<<<4096, 256, 0, stream>>>(x, xb, 8L * 64 * 256 * 256);

    // conv1: 64 -> 128, 3x3, pad 1.  Tile 128 couts x 128 px (4x2 waves)
    conv_wmma<64, 128, 576, 4, 2, true, 256, 256>
        <<<4096, 256, 0, stream>>>(xb, w1b, b1, h1p);
    bn_stats<<<128 * 64, 256, 0, stream>>>(h1p, sum1, sq1, 128, 65536, 8L * 65536, 64);
    bn_finalize<<<1, 128, 0, stream>>>(sum1, sq1, g1, be1, sc1, sh1, 128, 1.f / (8.f * 65536.f));
    bn_mish_bf16<<<8192, 256, 0, stream>>>(h1p, sc1, sh1, h1b, 128, 65536, 8L * 128 * 65536);

    // conv2: 128 -> 64, 3x3, pad 1.  Tile 64 couts x 256 px (2x4 waves)
    conv_wmma<128, 64, 1152, 2, 4, true, 256, 256>
        <<<2048, 256, 0, stream>>>(h1b, w2b, b2, h2p);
    bn_stats<<<64 * 64, 256, 0, stream>>>(h2p, sum2, sq2, 64, 65536, 8L * 65536, 64);
    bn_finalize<<<1, 64, 0, stream>>>(sum2, sq2, g2, be2, sc2, sh2, 64, 1.f / (8.f * 65536.f));

    // fused BN+mish+residual+Haar: cA -> d_out low half, bands -> bf16 buffer
    float* outLow  = (float*)d_out;
    float* outHigh = (float*)d_out + 8L * 64 * 128 * 128;
    fuse_residual_haar<<<8192, 256, 0, stream>>>(h2p, x, sc2, sh2, outLow, hib);

    // 1x1 conv: 192 -> 64 on (8,192,128,128).  Tile 64 couts x 256 px
    conv_wmma<192, 64, 192, 2, 4, false, 128, 128>
        <<<512, 256, 0, stream>>>(hib, whb, bh, hhp);
    bn_stats<<<64 * 64, 256, 0, stream>>>(hhp, sumh, sqh, 64, 16384, 8L * 16384, 64);
    bn_finalize<<<1, 64, 0, stream>>>(sumh, sqh, gh, beh, sch, shh, 64, 1.f / (8.f * 16384.f));
    bn_mish_f32<<<4096, 256, 0, stream>>>(hhp, sch, shh, outHigh, 64, 16384, 8L * 64 * 16384);
}
